// ReasonModule_30837865185800
// MI455X (gfx1250) — matrix operand order repaired and checked by reading, not compiled, not run on hardware
//
#include <hip/hip_runtime.h>
#include <hip/hip_bf16.h>

#define NB 1024      // batch segments
#define DD 256       // channels
#define KTOT 768     // 2D + D concat K
#define G4D 1024     // 4*D gate width
#define CH  128      // rows per LDS chunk in fused attention (128 KB tile)

typedef __attribute__((ext_vector_type(16))) __bf16 v16bf;
typedef __attribute__((ext_vector_type(8)))  float  v8f;
typedef __attribute__((ext_vector_type(4)))  unsigned int v4u;
typedef __attribute__((ext_vector_type(4)))  unsigned int u32x4;
typedef __attribute__((ext_vector_type(8)))  int i32x8;
typedef __attribute__((ext_vector_type(4)))  int i32x4;

union ABu { v16bf v; v4u u[2]; };

#if defined(__HIP_DEVICE_COMPILE__) && __has_builtin(__builtin_amdgcn_tensor_load_to_lds) && __has_builtin(__builtin_amdgcn_s_wait_tensorcnt)
#define USE_TDM 1
#else
#define USE_TDM 0
#endif

// ---------------- segment offsets (batch is sorted; compute once) ----------
__global__ void k_hist(const int* __restrict__ batch, int* __restrict__ counts, int n) {
    int i = blockIdx.x * blockDim.x + threadIdx.x;
    if (i < n) atomicAdd(&counts[batch[i]], 1);
}

__global__ void k_scan(const int* __restrict__ counts, int* __restrict__ off) {
    __shared__ int sh[NB];
    int tid = threadIdx.x;
    sh[tid] = counts[tid];
    __syncthreads();
    for (int st = 1; st < NB; st <<= 1) {
        int v = (tid >= st) ? sh[tid - st] : 0;
        __syncthreads();
        sh[tid] += v;
        __syncthreads();
    }
    if (tid == 0) off[0] = 0;
    off[tid + 1] = sh[tid];
}

// ---------------- one-time weight prep: Wt[k][n] = bf16(Wcat[n][k]) --------
__global__ void k_prepW(const float* __restrict__ W_ih, const float* __restrict__ W_hh,
                        __bf16* __restrict__ Wt) {
    int idx = blockIdx.x * blockDim.x + threadIdx.x;   // KTOT * G4D
    int k = idx >> 10;          // 0..767
    int n = idx & (G4D - 1);    // 0..1023
    float v = (k < 2 * DD) ? W_ih[(size_t)n * (2 * DD) + k]
                           : W_hh[(size_t)n * DD + (k - 2 * DD)];
    Wt[idx] = (__bf16)v;
}

// ---------------- per-step: pack A = bf16([q_star | h]) --------------------
__global__ void k_buildA(const float* __restrict__ qs, const float* __restrict__ h,
                         __bf16* __restrict__ A) {
    int idx = blockIdx.x * blockDim.x + threadIdx.x;   // NB * KTOT
    int b = idx / KTOT;
    int k = idx - b * KTOT;
    float v = (k < 2 * DD) ? qs[(size_t)b * (2 * DD) + k]
                           : h[(size_t)b * DD + (k - 2 * DD)];
    A[idx] = (__bf16)v;
}

// ---------------- WMMA GEMM: gates[NB x G4D] = A[NB x KTOT] * Wt ----------
__global__ __launch_bounds__(256) void k_gemm(const __bf16* __restrict__ A,
                                              const __bf16* __restrict__ Wt,
                                              float* __restrict__ gates) {
    int wv   = (blockIdx.x * blockDim.x + threadIdx.x) >> 5;  // global wave id
    int lane = threadIdx.x & 31;
    int tm = wv >> 6;           // 0..63 (row tiles)
    int tn = wv & 63;           // 0..63 (col tiles)

    int halfsel = lane >> 4;    // 0: lanes 0-15, 1: lanes 16-31
    int row = tm * 16 + (lane & 15);

    // A fragment (16-bit A 16x32 layout, ISA 7.12.2)
    const __bf16* arow = A + (size_t)row * KTOT + (halfsel << 3);

    v8f acc = {};
    for (int kb = 0; kb < KTOT / 32; ++kb) {
        ABu a, b;
        const __bf16* ap = arow + kb * 32;
        a.u[0] = *(const v4u*)(ap);        // K base+0..7
        a.u[1] = *(const v4u*)(ap + 16);   // K base+16..23
        // B fragment (32x16): lane = N column, 16 K values per lane (same K order as A)
        const __bf16* bp = Wt + (size_t)(kb * 32 + (halfsel << 3)) * G4D + tn * 16 + (lane & 15);
#pragma unroll
        for (int s = 0; s < 16; ++s)
            b.v[s] = bp[(size_t)((s < 8) ? s : s + 8) * G4D];
        acc = __builtin_amdgcn_wmma_f32_16x16x32_bf16(
            false, a.v, false, b.v, (short)0, acc, false, false);
    }

    int colb = tn * 16 + (lane & 15);
    int rowb = tm * 16 + (halfsel << 3);
#pragma unroll
    for (int v = 0; v < 8; ++v)
        gates[(size_t)(rowb + v) * G4D + colb] = acc[v];
}

// ---------------- LSTM pointwise: i,f,g,o -> h,c ---------------------------
__global__ void k_cell(const float* __restrict__ gates, const float* __restrict__ b_ih,
                       const float* __restrict__ b_hh, float* __restrict__ h,
                       float* __restrict__ c) {
    int idx = blockIdx.x * blockDim.x + threadIdx.x;  // NB*DD
    int b = idx >> 8, d = idx & (DD - 1);
    const float* g = gates + (size_t)b * G4D;
    float gi = g[d]            + b_ih[d]            + b_hh[d];
    float gf = g[DD + d]       + b_ih[DD + d]       + b_hh[DD + d];
    float gg = g[2 * DD + d]   + b_ih[2 * DD + d]   + b_hh[2 * DD + d];
    float go = g[3 * DD + d]   + b_ih[3 * DD + d]   + b_hh[3 * DD + d];
    float i_ = 1.f / (1.f + expf(-gi));
    float f_ = 1.f / (1.f + expf(-gf));
    float g_ = tanhf(gg);
    float o_ = 1.f / (1.f + expf(-go));
    float cn = f_ * c[idx] + i_ * g_;
    c[idx] = cn;
    h[idx] = o_ * tanhf(cn);
}

// ---------------- fused attention: single pass over x per step -------------
// One block per segment. Chunks of CH rows are DMA'd into LDS by the Tensor
// Data Mover (TENSORcnt), logits computed wave-per-row from LDS, online
// softmax (running max/sum with rescaling) and weighted accumulation.
__global__ __launch_bounds__(256) void k_attn_fused(const float* __restrict__ x,
                                                    const int* __restrict__ off,
                                                    const float* __restrict__ h,
                                                    float* __restrict__ qs_out) {
    __shared__ float xbuf[CH][DD];   // 128 KB tile (LDS up to 320 KB on CDNA5)
    __shared__ float esh[CH];
    __shared__ float red[256];

    int seg = blockIdx.x;
    int tid = threadIdx.x;
    int lane = tid & 31;
    int wv   = tid >> 5;
    int s = off[seg], t = off[seg + 1];

    // per-lane q fragment (q = h[seg]) for the row dots
    const float4* qrow = (const float4*)(h + (size_t)seg * DD);
    float4 q0 = qrow[lane * 2];
    float4 q1 = qrow[lane * 2 + 1];

    float m = -__builtin_inff();
    float ssum = 0.f;
    float racc = 0.f;

    for (int base = s; base < t; base += CH) {
        int rows = min(CH, t - base);

        // ---- stage rows [base, base+rows) of x into LDS ----
#if USE_TDM
        if (tid < 32) {
            unsigned lds_off = (unsigned)(unsigned long long)(&xbuf[0][0]);
            unsigned long long ga = (unsigned long long)(x + (size_t)base * DD);
            u32x4 g0;
            g0.x = 1u;                                        // count=1 (valid D#)
            g0.y = lds_off;                                   // lds_addr
            g0.z = (unsigned)(ga & 0xFFFFFFFFull);            // global_addr[31:0]
            g0.w = ((unsigned)(ga >> 32) & 0x01FFFFFFu)       // global_addr[56:32]
                   | (2u << 30);                              // type = 2 (image)
            i32x8 g1;
            g1.s0 = (int)(2u << 16);                          // data_size = 4 bytes
            g1.s1 = (int)((256u & 0xFFFFu) << 16);            // tensor_dim0[15:0]=256
            g1.s2 = (int)(((unsigned)rows & 0xFFFFu) << 16);  // dim0 hi=0 | tensor_dim1 lo=rows
            g1.s3 = (int)((((unsigned)rows >> 16) & 0xFFFFu)  // tensor_dim1 hi
                          | (256u << 16));                    // tile_dim0 = 256
            g1.s4 = (int)((unsigned)rows & 0xFFFFu);          // tile_dim1 = rows, tile_dim2=0
            g1.s5 = 256;                                      // tensor_dim0_stride = 256
            g1.s6 = 0;
            g1.s7 = 0;
            i32x4 z4 = {0, 0, 0, 0};
            i32x8 z8 = {0, 0, 0, 0, 0, 0, 0, 0};
            __builtin_amdgcn_tensor_load_to_lds(g0, g1, z4, z4, z8, 0);
            __builtin_amdgcn_s_wait_tensorcnt(0);
        }
#else
        {
            const float4* src = (const float4*)(x + (size_t)base * DD);
            float4* dst = (float4*)&xbuf[0][0];
            for (int idx = tid; idx < rows * (DD / 4); idx += 256)
                dst[idx] = src[idx];
        }
#endif
        __syncthreads();

        // ---- logits: wave-per-row dot from LDS, shfl reduce ----
        for (int j = wv; j < rows; j += 8) {
            const float4* xr = (const float4*)&xbuf[j][0];
            float4 a0 = xr[lane * 2];
            float4 a1 = xr[lane * 2 + 1];
            float d = a0.x * q0.x + a0.y * q0.y + a0.z * q0.z + a0.w * q0.w
                    + a1.x * q1.x + a1.y * q1.y + a1.z * q1.z + a1.w * q1.w;
#pragma unroll
            for (int mm = 16; mm > 0; mm >>= 1) d += __shfl_xor(d, mm, 32);
            if (lane == 0) esh[j] = d;
        }
        __syncthreads();

        // ---- chunk max (CH <= 256, single pass) ----
        red[tid] = (tid < rows) ? esh[tid] : -__builtin_inff();
        __syncthreads();
        for (int st = 128; st > 0; st >>= 1) {
            if (tid < st) red[tid] = fmaxf(red[tid], red[tid + st]);
            __syncthreads();
        }
        float cmax = red[0];
        float m_new = fmaxf(m, cmax);
        __syncthreads();

        // ---- p_j = exp(e_j - m_new) and chunk sum ----
        float p = 0.f;
        if (tid < rows) { p = expf(esh[tid] - m_new); esh[tid] = p; }
        red[tid] = p;
        __syncthreads();
        for (int st = 128; st > 0; st >>= 1) {
            if (tid < st) red[tid] += red[tid + st];
            __syncthreads();
        }
        float csum = red[0];
        __syncthreads();

        // ---- online rescale + accumulate (thread tid owns channel d) ----
        float scale = expf(m - m_new);
        ssum = ssum * scale + csum;
        float part = 0.f;
        for (int j = 0; j < rows; ++j) part += esh[j] * xbuf[j][tid];
        racc = racc * scale + part;
        m = m_new;
        __syncthreads();  // before next chunk overwrites xbuf/esh
    }

    qs_out[(size_t)seg * (2 * DD) + tid]      = h[(size_t)seg * DD + tid];
    qs_out[(size_t)seg * (2 * DD) + DD + tid] = racc / (ssum + 1e-16f);
}

// ---------------------------------------------------------------------------
extern "C" void kernel_launch(void* const* d_in, const int* in_sizes, int n_in,
                              void* d_out, int out_size, void* d_ws, size_t ws_size,
                              hipStream_t stream) {
    const float* x      = (const float*)d_in[0];
    const int*   batch  = (const int*)d_in[2];
    const float* qs_in  = (const float*)d_in[3];
    const float* W_ih   = (const float*)d_in[4];
    const float* W_hh   = (const float*)d_in[5];
    const float* b_ih   = (const float*)d_in[6];
    const float* b_hh   = (const float*)d_in[7];
    const int N = in_sizes[0] / DD;

    // workspace carve-out
    size_t off = 0;
    auto carve = [&](size_t bytes) {
        void* p = (char*)d_ws + off;
        off += (bytes + 255) & ~(size_t)255;
        return p;
    };
    float*  qsA    = (float*)carve((size_t)NB * 2 * DD * 4);
    float*  qsB    = (float*)carve((size_t)NB * 2 * DD * 4);
    float*  h      = (float*)carve((size_t)NB * DD * 4);
    float*  c      = (float*)carve((size_t)NB * DD * 4);
    __bf16* A      = (__bf16*)carve((size_t)NB * KTOT * 2);
    __bf16* Wt     = (__bf16*)carve((size_t)KTOT * G4D * 2);
    float*  gates  = (float*)carve((size_t)NB * G4D * 4);
    int*    counts = (int*)carve((size_t)NB * 4);
    int*    segoff = (int*)carve((size_t)(NB + 1) * 4);
    (void)ws_size; (void)n_in; (void)out_size;

    (void)hipMemsetAsync(counts, 0, (size_t)NB * 4, stream);
    (void)hipMemsetAsync(h, 0, (size_t)NB * DD * 4, stream);
    (void)hipMemsetAsync(c, 0, (size_t)NB * DD * 4, stream);
    (void)hipMemcpyAsync(qsA, qs_in, (size_t)NB * 2 * DD * 4, hipMemcpyDeviceToDevice, stream);

    k_hist<<<(N + 255) / 256, 256, 0, stream>>>(batch, counts, N);
    k_scan<<<1, NB, 0, stream>>>(counts, segoff);
    k_prepW<<<(KTOT * G4D) / 256, 256, 0, stream>>>(W_ih, W_hh, Wt);

    const float* qin = qsA;
    for (int step = 0; step < 3; ++step) {
        float* qout = (step == 2) ? (float*)d_out : ((step == 0) ? qsB : qsA);
        k_buildA<<<(NB * KTOT) / 256, 256, 0, stream>>>(qin, h, A);
        k_gemm<<<(64 * 64) / 8, 256, 0, stream>>>(A, Wt, gates);
        k_cell<<<(NB * DD) / 256, 256, 0, stream>>>(gates, b_ih, b_hh, h, c);
        k_attn_fused<<<NB, 256, 0, stream>>>(x, segoff, h, qout);
        qin = qout;
    }
}
